// OCRAttention_34376918237541
// MI455X (gfx1250) — matrix operand order repaired and checked by reading, not compiled, not run on hardware
//
#include <hip/hip_runtime.h>
#include <hip/hip_bf16.h>

// ---------------------------------------------------------------------------
// OCR attention forward for MI455X (gfx1250, wave32, WMMA + TDM/async-LDS).
//   Stage 1: QKV projection  (f16 WMMA, fp32 acc)  -> ws Q/K/V [B*H][S][64] f16
//   Stage 2: flash attention; K/V tiles staged via Tensor Data Mover
//            (tensor_load_to_lds + s_wait_tensorcnt)
//   Stage 3: output projection; ctx tile staged via global_load_async_to_lds
// ocr_bias is a per-head constant over all (q,k): softmax is shift-invariant,
// so it cancels exactly and is skipped.
// ---------------------------------------------------------------------------

typedef __attribute__((ext_vector_type(16))) _Float16 v16h;
typedef __attribute__((ext_vector_type(8)))  float    v8f;

#define D_MODEL 1024
#define N_HEADS 16
#define D_K     64
#define BATCH   2
#define SEQ     2048
#define M_TOTAL (BATCH * SEQ)                 // 4096 rows of x / ctx
#define SECT    ((size_t)M_TOTAL * D_MODEL)   // elements per Q/K/V section

#if __has_builtin(__builtin_amdgcn_tensor_load_to_lds)
#define HAVE_TDM 1
#endif
#if __has_builtin(__builtin_amdgcn_global_load_async_to_lds_b128)
#define HAVE_ASYNC 1
#endif

// ---- address-space helpers ------------------------------------------------
typedef int i32x4 __attribute__((ext_vector_type(4)));
typedef int i32x8 __attribute__((ext_vector_type(8)));
typedef unsigned int u32x4 __attribute__((ext_vector_type(4)));

typedef __attribute__((address_space(1))) i32x4 glb_i32x4;
typedef __attribute__((address_space(3))) i32x4 lds_i32x4;

static __device__ __forceinline__ unsigned lds_offset_of(const void* p) {
    // generic LDS address: low 32 bits are the workgroup-relative LDS offset
    return (unsigned)(size_t)p;
}
static __device__ __forceinline__ glb_i32x4* to_glb16(const void* p) {
    return (glb_i32x4*)(size_t)p;
}
static __device__ __forceinline__ lds_i32x4* to_lds16(void* p) {
    return (lds_i32x4*)(unsigned)(size_t)p;
}

static __device__ __forceinline__ void wait_asynccnt0() {
#if __has_builtin(__builtin_amdgcn_s_wait_asynccnt)
    __builtin_amdgcn_s_wait_asynccnt(0);
#else
    asm volatile("s_wait_asynccnt 0" ::: "memory");
#endif
}

#ifdef HAVE_TDM
// TDM: DMA a rows x cols f16 tile (contiguous, row stride == cols) into LDS.
// D# per CDNA5 ISA 8.3/8.4: group0 = {count/type, lds_addr, global_addr},
// group1 = {data_size, tensor dims, tile dims, strides}; higher groups unused
// for a 2D tile and passed as zeros (6-arg clang-23 builtin form).
static __device__ __forceinline__ void
tdm_load_tile_f16(const _Float16* gsrc, unsigned lds_byte_off, int rows, int cols) {
    const unsigned long long ga = (unsigned long long)(size_t)gsrc;
    u32x4 g0;
    g0[0] = 1u;                                            // count=1 (valid D#)
    g0[1] = lds_byte_off;                                  // lds_addr
    g0[2] = (unsigned)(ga & 0xffffffffull);                // global_addr[31:0]
    g0[3] = (unsigned)((ga >> 32) & 0x1ffffffull)          // global_addr[56:32]
          | (2u << 30);                                    // type=2 ("image")
    i32x8 g1;
    g1[0] = (1 << 16);                                     // data_size=1 -> 2 bytes
    g1[1] = (cols & 0xffff) << 16;                         // tensor_dim0[15:0]
    g1[2] = (rows & 0xffff) << 16;                         // tensor_dim1[15:0]
    g1[3] = (cols & 0xffff) << 16;                         // tile_dim0
    g1[4] = (rows & 0xffff);                               // tile_dim1, tile_dim2=0
    g1[5] = cols;                                          // tensor_dim0_stride lo
    g1[6] = 0;                                             // stride hi / dim1_stride
    g1[7] = 0;
    const i32x4 z4 = {0, 0, 0, 0};                         // groups 2/3: unused (2D)
    const i32x8 z8 = {0, 0, 0, 0, 0, 0, 0, 0};
    __builtin_amdgcn_tensor_load_to_lds(g0, g1, z4, z4, z8, 0);
}
#endif

// ---- WMMA fragment loaders (contiguous k order k = 16*(lane/16) + e, A and B alike) --------------------
// A fragment, 16x32 f16 (M rows x K): lane L<16 -> M=L, K base 0;
// lane L>=16 -> M=L-16, K base 16.  element e: k = base + e.
static __device__ __forceinline__ v16h
load_a_frag(const _Float16* src, int row0, int col0, int ld) {
    const int l  = threadIdx.x & 31;
    const int m  = row0 + (l & 15);
    const int kb = (l < 16) ? 0 : 16;
    const _Float16* p = src + (size_t)m * ld + col0;
    v16h a;
#pragma unroll
    for (int e = 0; e < 16; ++e) a[e] = p[kb + e];
    return a;
}

// A fragment of a transposed operand: LDS tile stored [k][m] (A[m][k]=src[k][m]).
static __device__ __forceinline__ v16h
load_a_frag_t(const _Float16* src, int row0, int col0, int ld) {
    const int l  = threadIdx.x & 31;
    const int m  = row0 + (l & 15);
    const int kb = (l < 16) ? 0 : 16;
    const _Float16* p = src + m;
    v16h a;
#pragma unroll
    for (int e = 0; e < 16; ++e) {
        const int k = col0 + kb + e;
        a[e] = p[(size_t)k * ld];
    }
    return a;
}

// B fragment, 32x16 f16 from an LDS tile stored [n][k] (B[k][n]=src[n][k]):
// lane L -> N = L&15; lanes<16 hold K 0..15, lanes>=16 hold K 16..31.
static __device__ __forceinline__ v16h
load_b_frag_nk(const _Float16* src, int col0, int ld) {
    const int l  = threadIdx.x & 31;
    const int n  = col0 + (l & 15);
    const int kb = (l < 16) ? 0 : 16;
    const _Float16* p = src + (size_t)n * ld + kb;
    v16h b;
#pragma unroll
    for (int e = 0; e < 16; ++e) b[e] = p[e];
    return b;
}

// ---------------------------------------------------------------------------
// Stage 1: Q/K/V = x @ W^T + b, written head-major [b*H+h][s][d_k] as f16.
// (fp32 inputs must be converted, so staging stays synchronous here.)
// ---------------------------------------------------------------------------
__global__ __launch_bounds__(256) void qkv_proj_kernel(
    const float* __restrict__ x,
    const float* __restrict__ wq, const float* __restrict__ bq,
    const float* __restrict__ wk, const float* __restrict__ bk,
    const float* __restrict__ wv, const float* __restrict__ bv,
    _Float16* __restrict__ qkv)   // [3][SECT]
{
    const int which = blockIdx.z;
    const float* w    = (which == 0) ? wq : (which == 1) ? wk : wv;
    const float* bias = (which == 0) ? bq : (which == 1) ? bk : bv;
    _Float16* outb = qkv + (size_t)which * SECT;

    const int m0 = blockIdx.x * 64;
    const int n0 = blockIdx.y * 64;

    __shared__ _Float16 sA[64][34];
    __shared__ _Float16 sW[64][34];

    const int tid  = threadIdx.x;
    const int wid  = tid >> 5;
    const int msub = wid >> 1;
    const int nsub = wid & 1;
    const int lane = tid & 31;

    v8f cacc[2];
    {
        const int nc = n0 + nsub * 32 + (lane & 15);
        const float b0v = bias[nc];
        const float b1v = bias[nc + 16];
#pragma unroll
        for (int r = 0; r < 8; ++r) { cacc[0][r] = b0v; cacc[1][r] = b1v; }
    }

    for (int kk = 0; kk < D_MODEL; kk += 32) {
#pragma unroll
        for (int i = 0; i < 8; ++i) {
            const int e = i * 256 + tid;
            const int r = e >> 5, c = e & 31;
            sA[r][c] = (_Float16)x[(size_t)(m0 + r) * D_MODEL + kk + c];
            sW[r][c] = (_Float16)w[(size_t)(n0 + r) * D_MODEL + kk + c];
        }
        if (kk + 32 < D_MODEL) {
            __builtin_prefetch(&x[(size_t)(m0 + (tid >> 3)) * D_MODEL + kk + 32], 0, 1);
            __builtin_prefetch(&w[(size_t)(n0 + (tid >> 3)) * D_MODEL + kk + 32], 0, 1);
        }
        __syncthreads();

        const v16h a  = load_a_frag(&sA[0][0], msub * 16, 0, 34);
        const v16h b0 = load_b_frag_nk(&sW[0][0], nsub * 32 + 0, 34);
        const v16h b1 = load_b_frag_nk(&sW[0][0], nsub * 32 + 16, 34);
        cacc[0] = __builtin_amdgcn_wmma_f32_16x16x32_f16(false, a, false, b0,
                                                         (short)0, cacc[0], false, false);
        cacc[1] = __builtin_amdgcn_wmma_f32_16x16x32_f16(false, a, false, b1,
                                                         (short)0, cacc[1], false, false);
        __syncthreads();
    }

    const int nl = lane & 15;
#pragma unroll
    for (int t = 0; t < 2; ++t) {
        const int n  = n0 + nsub * 32 + t * 16 + nl;
        const int hh = n >> 6, dd = n & (D_K - 1);
#pragma unroll
        for (int r = 0; r < 8; ++r) {
            const int m  = m0 + msub * 16 + r + ((lane < 16) ? 0 : 8);
            const int bb = m >> 11, ss = m & (SEQ - 1);
            outb[(((size_t)(bb * N_HEADS + hh) * SEQ) + ss) * D_K + dd] =
                (_Float16)cacc[t][r];
        }
    }
}

// ---------------------------------------------------------------------------
// Stage 2: flash attention per (b,h); 8 waves x 16 queries. K/V tiles (32x64
// f16, verbatim) arrive via TDM; scores^T = K @ Q^T and ctx^T = V^T @ P^T keep
// the query in the lane dimension -> per-lane scalar online softmax.
// ---------------------------------------------------------------------------
__global__ __launch_bounds__(256) void attn_kernel(
    const _Float16* __restrict__ qkv, _Float16* __restrict__ ctx /* [B][S][D] f16 */)
{
    const int bh = blockIdx.y;
    const int b  = bh >> 4, h = bh & (N_HEADS - 1);
    const int tid = threadIdx.x, wid = tid >> 5, lane = tid & 31;
    const int qw = blockIdx.x * 128 + wid * 16;

    const _Float16* Qp = qkv + (size_t)bh * SEQ * D_K;
    const _Float16* Kp = Qp + SECT;
    const _Float16* Vp = Qp + 2 * SECT;

    v16h qb[2];
    {
        const int n  = lane & 15;
        const int kb = (lane < 16) ? 0 : 16;
        const _Float16* p = Qp + (size_t)(qw + n) * D_K;
#pragma unroll
        for (int f = 0; f < 2; ++f)
#pragma unroll
            for (int e = 0; e < 16; ++e) qb[f][e] = p[f * 32 + kb + e];
    }

    __shared__ _Float16 sK[32][64];   // [key][d], packed (DMA-friendly)
    __shared__ _Float16 sV[32][64];   // [key][d], packed

    float row_max = -3.0e38f, row_sum = 0.0f;
    v8f o[4];
#pragma unroll
    for (int i = 0; i < 4; ++i)
#pragma unroll
        for (int r = 0; r < 8; ++r) o[i][r] = 0.0f;

    for (int kv0 = 0; kv0 < SEQ; kv0 += 32) {
#ifdef HAVE_TDM
        // one wave drives the Tensor Data Mover for both tiles (EXEC-agnostic)
        if (wid == 0) {
            tdm_load_tile_f16(Kp + (size_t)kv0 * D_K, lds_offset_of(&sK[0][0]), 32, 64);
            tdm_load_tile_f16(Vp + (size_t)kv0 * D_K, lds_offset_of(&sV[0][0]), 32, 64);
            __builtin_amdgcn_s_wait_tensorcnt(0);
        }
        __syncthreads();
#elif defined(HAVE_ASYNC)
        {   // 256 threads x one 16B chunk per tile (rows are 128B, 16B-aligned)
            const int key = tid >> 3, d8 = (tid & 7) * 8;
            __builtin_amdgcn_global_load_async_to_lds_b128(
                to_glb16(Kp + (size_t)(kv0 + key) * D_K + d8), to_lds16(&sK[key][d8]), 0, 0);
            __builtin_amdgcn_global_load_async_to_lds_b128(
                to_glb16(Vp + (size_t)(kv0 + key) * D_K + d8), to_lds16(&sV[key][d8]), 0, 0);
            wait_asynccnt0();
        }
        __syncthreads();
#else
#pragma unroll
        for (int i = 0; i < 8; ++i) {
            const int e = i * 256 + tid;
            const int key = e >> 6, d = e & (D_K - 1);
            sK[key][d] = Kp[(size_t)(kv0 + key) * D_K + d];
            sV[key][d] = Vp[(size_t)(kv0 + key) * D_K + d];
        }
        __syncthreads();
#endif

        // scores^T: st[t] = K[16t..16t+15] x Q^T over d_k halves
        v8f st[2];
#pragma unroll
        for (int t = 0; t < 2; ++t) {
            const v16h a0 = load_a_frag(&sK[0][0], 16 * t, 0, 64);
            const v16h a1 = load_a_frag(&sK[0][0], 16 * t, 32, 64);
            v8f z = {};
            z = __builtin_amdgcn_wmma_f32_16x16x32_f16(false, a0, false, qb[0],
                                                       (short)0, z, false, false);
            st[t] = __builtin_amdgcn_wmma_f32_16x16x32_f16(false, a1, false, qb[1],
                                                           (short)0, z, false, false);
        }

#pragma unroll
        for (int t = 0; t < 2; ++t)
#pragma unroll
            for (int r = 0; r < 8; ++r) st[t][r] *= 0.125f;   // 1/sqrt(64)

        float tm = -3.0e38f;
#pragma unroll
        for (int t = 0; t < 2; ++t)
#pragma unroll
            for (int r = 0; r < 8; ++r) tm = fmaxf(tm, st[t][r]);
        tm = fmaxf(tm, __shfl_xor(tm, 16, 32));
        const float new_max = fmaxf(row_max, tm);
        const float alpha   = __expf(row_max - new_max);

        float tsum = 0.0f;
#pragma unroll
        for (int t = 0; t < 2; ++t)
#pragma unroll
            for (int r = 0; r < 8; ++r) {
                const float pex = __expf(st[t][r] - new_max);
                st[t][r] = pex;
                tsum += pex;
            }
        tsum += __shfl_xor(tsum, 16, 32);
        row_sum = row_sum * alpha + tsum;
        row_max = new_max;
#pragma unroll
        for (int i = 0; i < 4; ++i)
#pragma unroll
            for (int r = 0; r < 8; ++r) o[i][r] *= alpha;

        // repack P^T D-fragments into a 32x16 B-fragment via half-swap shuffles
        v16h pb;
#pragma unroll
        for (int i = 0; i < 8; ++i) {
            const float own0 = st[0][i], own1 = st[1][i];
            const float oth0 = __shfl_xor(own0, 16, 32);
            const float oth1 = __shfl_xor(own1, 16, 32);
            pb[i]     = (_Float16)((lane < 16) ? own0 : oth1);
            pb[i + 8] = (_Float16)((lane < 16) ? oth0 : own1);
        }

        // ctx^T += V^T(16 d-rows x 32 keys) @ P^T; V^T read via transposed frags
#pragma unroll
        for (int dt = 0; dt < 4; ++dt) {
            const v16h a = load_a_frag_t(&sV[0][0], 16 * dt, 0, 64);
            o[dt] = __builtin_amdgcn_wmma_f32_16x16x32_f16(false, a, false, pb,
                                                           (short)0, o[dt], false, false);
        }
        __syncthreads();
    }

    const float inv = 1.0f / row_sum;
    const int q = qw + (lane & 15);
#pragma unroll
    for (int dt = 0; dt < 4; ++dt)
#pragma unroll
        for (int r = 0; r < 8; ++r) {
            const int d = 16 * dt + r + ((lane < 16) ? 0 : 8);
            ctx[((size_t)b * SEQ + q) * D_MODEL + h * D_K + d] =
                (_Float16)(o[dt][r] * inv);
        }
}

// ---------------------------------------------------------------------------
// Stage 3: out = ctx @ wo^T + bo (fp32 result). ctx tile (already f16) is
// async-copied to LDS, overlapped with fp32->f16 staging of the weight tile.
// ---------------------------------------------------------------------------
__global__ __launch_bounds__(256) void out_proj_kernel(
    const _Float16* __restrict__ ctx, const float* __restrict__ wo,
    const float* __restrict__ bo, float* __restrict__ out)
{
    const int m0 = blockIdx.x * 64;
    const int n0 = blockIdx.y * 64;

    __shared__ _Float16 sA[64][40];   // pad = 8 halfs -> rows stay 16B-aligned
    __shared__ _Float16 sW[64][34];

    const int tid  = threadIdx.x;
    const int wid  = tid >> 5;
    const int msub = wid >> 1;
    const int nsub = wid & 1;
    const int lane = tid & 31;

    v8f cacc[2];
    {
        const int nc = n0 + nsub * 32 + (lane & 15);
        const float b0v = bo[nc];
        const float b1v = bo[nc + 16];
#pragma unroll
        for (int r = 0; r < 8; ++r) { cacc[0][r] = b0v; cacc[1][r] = b1v; }
    }

    for (int kk = 0; kk < D_MODEL; kk += 32) {
#ifdef HAVE_ASYNC
        {   // ctx tile: 64 rows x 32 halfs = 256 x 16B chunks, one per thread
            const int r = tid >> 2, c8 = (tid & 3) * 8;
            __builtin_amdgcn_global_load_async_to_lds_b128(
                to_glb16(ctx + (size_t)(m0 + r) * D_MODEL + kk + c8),
                to_lds16(&sA[r][c8]), 0, 0);
        }
#pragma unroll
        for (int i = 0; i < 8; ++i) {   // weight tile needs fp32->f16 convert
            const int e = i * 256 + tid;
            const int r = e >> 5, c = e & 31;
            sW[r][c] = (_Float16)wo[(size_t)(n0 + r) * D_MODEL + kk + c];
        }
        wait_asynccnt0();
#else
#pragma unroll
        for (int i = 0; i < 8; ++i) {
            const int e = i * 256 + tid;
            const int r = e >> 5, c = e & 31;
            sA[r][c] = ctx[(size_t)(m0 + r) * D_MODEL + kk + c];
            sW[r][c] = (_Float16)wo[(size_t)(n0 + r) * D_MODEL + kk + c];
        }
#endif
        if (kk + 32 < D_MODEL) {
            __builtin_prefetch(&ctx[(size_t)(m0 + (tid >> 3)) * D_MODEL + kk + 32], 0, 1);
            __builtin_prefetch(&wo[(size_t)(n0 + (tid >> 3)) * D_MODEL + kk + 32], 0, 1);
        }
        __syncthreads();

        const v16h a  = load_a_frag(&sA[0][0], msub * 16, 0, 40);
        const v16h b0 = load_b_frag_nk(&sW[0][0], nsub * 32 + 0, 34);
        const v16h b1 = load_b_frag_nk(&sW[0][0], nsub * 32 + 16, 34);
        cacc[0] = __builtin_amdgcn_wmma_f32_16x16x32_f16(false, a, false, b0,
                                                         (short)0, cacc[0], false, false);
        cacc[1] = __builtin_amdgcn_wmma_f32_16x16x32_f16(false, a, false, b1,
                                                         (short)0, cacc[1], false, false);
        __syncthreads();
    }

    const int nl = lane & 15;
#pragma unroll
    for (int t = 0; t < 2; ++t) {
        const int n = n0 + nsub * 32 + t * 16 + nl;
#pragma unroll
        for (int r = 0; r < 8; ++r) {
            const int m = m0 + msub * 16 + r + ((lane < 16) ? 0 : 8);
            out[(size_t)m * D_MODEL + n] = cacc[t][r];
        }
    }
}

// ---------------------------------------------------------------------------
extern "C" void kernel_launch(void* const* d_in, const int* in_sizes, int n_in,
                              void* d_out, int out_size, void* d_ws, size_t ws_size,
                              hipStream_t stream) {
    (void)in_sizes; (void)n_in; (void)out_size; (void)ws_size;

    const float* x  = (const float*)d_in[0];
    const float* wq = (const float*)d_in[1];
    const float* bq = (const float*)d_in[2];
    const float* wk = (const float*)d_in[3];
    const float* bk = (const float*)d_in[4];
    const float* wv = (const float*)d_in[5];
    const float* bv = (const float*)d_in[6];
    const float* wo = (const float*)d_in[7];
    const float* bo = (const float*)d_in[8];
    // d_in[9] = ocr_bias: per-head constant over all (q,k) -> cancels in softmax.

    _Float16* qkv = (_Float16*)d_ws;          // 3 * SECT f16 (24 MB)
    _Float16* ctx = qkv + 3 * SECT;           // SECT f16 (8 MB)

    qkv_proj_kernel<<<dim3(M_TOTAL / 64, D_MODEL / 64, 3), 256, 0, stream>>>(
        x, wq, bq, wk, bk, wv, bv, qkv);
    attn_kernel<<<dim3(SEQ / 128, BATCH * N_HEADS), 256, 0, stream>>>(qkv, ctx);
    out_proj_kernel<<<dim3(M_TOTAL / 64, D_MODEL / 64), 256, 0, stream>>>(
        ctx, wo, bo, (float*)d_out);
}
